// LlamaPerformerAttention_49237505081854
// MI455X (gfx1250) — compile-verified, matrix operand order checked by reading, not run on hardware
//
#include <hip/hip_runtime.h>

// ---------------- CDNA5 (gfx1250) wave32 WMMA performer-attention pipeline ----

typedef __bf16 v16bf __attribute__((ext_vector_type(16)));
typedef __bf16 v8bf  __attribute__((ext_vector_type(8)));
typedef float  v8f   __attribute__((ext_vector_type(8)));
typedef float  f32x4 __attribute__((ext_vector_type(4)));
typedef unsigned int   u32x4 __attribute__((ext_vector_type(4)));
typedef unsigned short u16x4 __attribute__((ext_vector_type(4)));

#define BB   2
#define SS   2048
#define HIDD 2048
#define NHH  32
#define NKVV 8
#define DD   64
#define MF   256
#define CH   128
#define NC   16
#define ROWS  (BB*SS)        // 4096
#define QROWS (ROWS*NHH)     // 131072
#define KROWS (ROWS*NKVV)    // 32768
#define NZ    (BB*NHH*NC)    // 1024 (b,h,chunk) batches
#define NZK   (BB*NKVV*NC)   // 256  (b,kvh,chunk) batches
#define EPSF  1e-6f
#define LDSW  40             // LDS row stride in halves (80B, 16B aligned)

static __device__ __forceinline__ __bf16 f2bf(float f) {
  union { float f; unsigned u; } a; a.f = f;
  unsigned r = a.u + 0x7FFFu + ((a.u >> 16) & 1u);
  union { unsigned short s; __bf16 h; } b; b.s = (unsigned short)(r >> 16);
  return b.h;
}
static __device__ __forceinline__ float bf2f(__bf16 h) {
  union { unsigned short s; __bf16 h; } b; b.h = h;
  union { unsigned u; float f; } a; a.u = ((unsigned)b.s) << 16;
  return a.f;
}

// CDNA5 async global->LDS copy, 16B per lane, tracked by ASYNCcnt.
static __device__ __forceinline__ void async_copy_b128(__bf16* ldst, const __bf16* gsrc) {
  unsigned lds = (unsigned)(unsigned long long)(void*)ldst;  // addr[31:0] = LDS offset
  asm volatile("global_load_async_to_lds_b128 %0, %1, off"
               :: "v"(lds), "v"((unsigned long long)gsrc)
               : "memory");
}
static __device__ __forceinline__ void wait_async() {
  asm volatile("s_wait_asynccnt 0x0" ::: "memory");
}

// ---------------------------------------------------------------- convert f32->bf16 (x4)
__global__ __launch_bounds__(256) void cvt_f32_bf16(const float* __restrict__ in,
                                                    __bf16* __restrict__ out,
                                                    long long n4) {
  long long i = (long long)blockIdx.x * blockDim.x + threadIdx.x;
  long long stride = (long long)gridDim.x * blockDim.x;
  for (; i < n4; i += stride) {
    f32x4 v = *(const f32x4*)(in + i * 4);
    union { u16x4 v; __bf16 e[4]; } o;
#pragma unroll
    for (int j = 0; j < 4; ++j) o.e[j] = f2bf(v[j]);
    *(u16x4*)(out + i * 4) = o.v;
  }
}

// ---------------------------------------------------------------- generic WMMA GEMM
// C[M,N] = sum_k Aelem(m,k)*Belem(n,k)
//   Aelem(m,k) = TRANS_A ? A[k*lda+m] : A[m*lda+k]
//   Belem(n,k) = TRANS_B ? B[k*ldb+n] : B[n*ldb+k]
// Block tile 128 x (32*CF), 8 waves (wave32), each wave 32 x (16*CF) = 2xCF wmma frags.
// Double-buffered LDS, async global->LDS staging, one barrier per k-step.
// ALL launches use M%128==0, N%(32*CF)==0, K%32==0 -> no bounds checks.
template <bool TRANS_A, bool TRANS_B, int CF>
__global__ __launch_bounds__(256) void gemm_bf16(const __bf16* __restrict__ A,
                                                 const __bf16* __restrict__ B,
                                                 float* __restrict__ C,
                                                 int M, int N, int K,
                                                 int lda, int ldb, int ldc, int mode) {
  constexpr int BN = 32 * CF;
  const int z = blockIdx.z;
  if (mode == 1) {                     // kv state: z=((b*8+kvh)*16+c)
    int c = z & 15, kvh = (z >> 4) & 7, b = z >> 7;
    A += ((long long)(b * SS + c * CH) * NKVV + kvh) * MF;        // phi_k  [t,m] (T)
    B += (long long)(b * SS + c * CH) * (NKVV * DD) + kvh * DD;   // v_bf16 [t,d] (T)
    C += (long long)z * (MF * DD);
  } else if (mode == 2) {              // inter: z=((b*32+h)*16+c)
    int c = z & 15, h = (z >> 4) & 31, b = z >> 9;
    A += ((long long)(b * SS + c * CH) * NHH + h) * MF;           // phi_q [s,m] (N)
    B += (long long)((b * NKVV + (h >> 2)) * NC + c) * (MF * DD); // kv_pre [m,d] (T)
    C += (long long)z * (CH * DD);
  } else if (mode == 3) {              // scores: z=((b*32+h)*16+c)
    int c = z & 15, h = (z >> 4) & 31, b = z >> 9;
    A += ((long long)(b * SS + c * CH) * NHH + h) * MF;           // phi_q (N)
    B += ((long long)(b * SS + c * CH) * NKVV + (h >> 2)) * MF;   // phi_k [t,m] (N)
    C += (long long)z * (CH * CH);
  } else if (mode == 4) {              // intra: z=((b*32+h)*16+c)
    int c = z & 15, h = (z >> 4) & 31, b = z >> 9;
    A += (long long)z * (CH * CH);                                // masked scores (N)
    B += (long long)(b * SS + c * CH) * (NKVV * DD) + (h >> 2) * DD; // v [t,d] (T)
    C += (long long)z * (CH * DD);
  }

  const int row0 = blockIdx.y * 128;
  const int col0 = blockIdx.x * BN;
  const int tid = threadIdx.x;
  const int lane = tid & 31;
  const int wave = tid >> 5;
  const int wm = wave >> 1;       // 0..3 -> 32-row slab
  const int wn = wave & 1;        // 0..1 -> (16*CF)-col slab
  const int half = lane >> 4;
  const int l15 = lane & 15;

  __shared__ __bf16 As[2][128 * LDSW];
  __shared__ __bf16 Bs[2][BN * LDSW];

  // per-thread staging coordinates (constant over k-steps)
  int a_m, a_k, b_n, b_k;
  if (!TRANS_A) { a_m = tid >> 1; a_k = (tid & 1) * 16; }   // 16 contiguous k halves
  else          { a_m = (tid & 7) * 16; a_k = tid >> 3; }   // 16 contiguous m, fixed k
  if (CF == 4) {
    if (!TRANS_B) { b_n = tid >> 1; b_k = (tid & 1) * 16; }
    else          { b_n = (tid & 7) * 16; b_k = tid >> 3; }
  } else {
    if (!TRANS_B) { b_n = tid >> 2; b_k = (tid & 3) * 8; }
    else          { b_n = (tid & 7) * 8; b_k = tid >> 3; }
  }

  auto stage = [&](int k0, int buf) {
    __bf16* as = As[buf];
    __bf16* bs = Bs[buf];
    // ---- A tile 128x32
    if (!TRANS_A) {
      const __bf16* gp = A + (long long)(row0 + a_m) * lda + (k0 + a_k);
      async_copy_b128(&as[a_m * LDSW + a_k], gp);
      async_copy_b128(&as[a_m * LDSW + a_k + 8], gp + 8);
      __builtin_prefetch(gp + 32, 0, 3);
    } else {
      const __bf16* gp = A + (long long)(k0 + a_k) * lda + (row0 + a_m);
      union { u32x4 v; __bf16 e[8]; } w0, w1;
      w0.v = *(const u32x4*)gp;
      w1.v = *(const u32x4*)(gp + 8);
#pragma unroll
      for (int j = 0; j < 8; ++j) as[(a_m + j) * LDSW + a_k] = w0.e[j];
#pragma unroll
      for (int j = 0; j < 8; ++j) as[(a_m + 8 + j) * LDSW + a_k] = w1.e[j];
      __builtin_prefetch(gp + (long long)32 * lda, 0, 3);
    }
    // ---- B tile BNx32
    if (CF == 4) {
      if (!TRANS_B) {
        const __bf16* gp = B + (long long)(col0 + b_n) * ldb + (k0 + b_k);
        async_copy_b128(&bs[b_n * LDSW + b_k], gp);
        async_copy_b128(&bs[b_n * LDSW + b_k + 8], gp + 8);
        __builtin_prefetch(gp + 32, 0, 3);
      } else {
        const __bf16* gp = B + (long long)(k0 + b_k) * ldb + (col0 + b_n);
        union { u32x4 v; __bf16 e[8]; } w0, w1;
        w0.v = *(const u32x4*)gp;
        w1.v = *(const u32x4*)(gp + 8);
#pragma unroll
        for (int j = 0; j < 8; ++j) bs[(b_n + j) * LDSW + b_k] = w0.e[j];
#pragma unroll
        for (int j = 0; j < 8; ++j) bs[(b_n + 8 + j) * LDSW + b_k] = w1.e[j];
        __builtin_prefetch(gp + (long long)32 * ldb, 0, 3);
      }
    } else {
      if (!TRANS_B) {
        const __bf16* gp = B + (long long)(col0 + b_n) * ldb + (k0 + b_k);
        async_copy_b128(&bs[b_n * LDSW + b_k], gp);
        __builtin_prefetch(gp + 32, 0, 3);
      } else {
        const __bf16* gp = B + (long long)(k0 + b_k) * ldb + (col0 + b_n);
        union { u32x4 v; __bf16 e[8]; } w;
        w.v = *(const u32x4*)gp;
#pragma unroll
        for (int j = 0; j < 8; ++j) bs[(b_n + j) * LDSW + b_k] = w.e[j];
        __builtin_prefetch(gp + (long long)32 * ldb, 0, 3);
      }
    }
  };

  union F8 { v8f v; float e[8]; };
  union F16 { v16bf v; v8bf h[2]; };
  F8 acc[2][CF];
#pragma unroll
  for (int r = 0; r < 2; ++r)
#pragma unroll
    for (int c = 0; c < CF; ++c) acc[r][c].v = (v8f)0.0f;

  auto compute = [&](int cur) {
    const __bf16* as = As[cur];
    const __bf16* bs = Bs[cur];
    // fragments per ISA 16-bit layout: lane(0..15)=row/col, half=lane>>4,
    // elem e -> K = 16*(e>>3) + 8*half + (e&7)  => two aligned ds_load_b128 each
    F16 af[2], bfr[CF];
#pragma unroll
    for (int r = 0; r < 2; ++r) {
      int m = wm * 32 + r * 16 + l15;
      af[r].h[0] = *(const v8bf*)(&as[m * LDSW + half * 8]);
      af[r].h[1] = *(const v8bf*)(&as[m * LDSW + 16 + half * 8]);
    }
#pragma unroll
    for (int c = 0; c < CF; ++c) {
      int n = wn * (16 * CF) + c * 16 + l15;
      bfr[c].h[0] = *(const v8bf*)(&bs[n * LDSW + half * 8]);
      bfr[c].h[1] = *(const v8bf*)(&bs[n * LDSW + 16 + half * 8]);
    }
#pragma unroll
    for (int r = 0; r < 2; ++r)
#pragma unroll
      for (int c = 0; c < CF; ++c)
        acc[r][c].v = __builtin_amdgcn_wmma_f32_16x16x32_bf16(
            false, af[r].v, false, bfr[c].v, (short)0, acc[r][c].v, false, false);
  };

  // software pipeline: stage(s+1) overlaps compute(s); one barrier per step
  stage(0, 0);
  wait_async();
  __syncthreads();
  const int nstep = K >> 5;
  int s = 0;
  for (; s < nstep - 1; ++s) {
    stage((s + 1) * 32, (s & 1) ^ 1);
    compute(s & 1);
    wait_async();
    __syncthreads();
  }
  compute(s & 1);

  // epilogue: D frag lane(0..15)=N, half selects M base (+8), vgpr v = row offset
#pragma unroll
  for (int r = 0; r < 2; ++r)
#pragma unroll
    for (int c = 0; c < CF; ++c) {
      int n = col0 + wn * (16 * CF) + c * 16 + l15;
#pragma unroll
      for (int v = 0; v < 8; ++v) {
        int m = row0 + wm * 32 + r * 16 + half * 8 + v;
        C[(long long)m * ldc + n] = acc[r][c].e[v];
      }
    }
}

// ------------------------------------------------- RoPE + scale + bf16 + 0.5*||x||^2
// one wave per (row=b*S+s, head); lane d handles pair (d, d+32)
__global__ __launch_bounds__(256) void rope_feat(const float* __restrict__ x,
                                                 const float* __restrict__ cosb,
                                                 const float* __restrict__ sinb,
                                                 __bf16* __restrict__ xf,
                                                 float* __restrict__ sq,
                                                 int rows, int H) {
  int gw = (blockIdx.x * blockDim.x + threadIdx.x) >> 5;
  int lane = threadIdx.x & 31;
  if (gw >= rows * H) return;
  int row = gw / H, h = gw % H;
  const float* xp = x + (long long)row * H * DD + h * DD;
  float x1 = xp[lane], x2 = xp[lane + 32];
  long long cb = (long long)row * DD;
  float c1 = cosb[cb + lane], s1 = sinb[cb + lane];
  float c2 = cosb[cb + lane + 32], s2 = sinb[cb + lane + 32];
  const float dn = 0.35355339059327373f;  // 64^(-1/4)
  float r1 = (x1 * c1 - x2 * s1) * dn;
  float r2 = (x2 * c2 + x1 * s2) * dn;
  __bf16* op = xf + (long long)row * H * DD + h * DD;
  op[lane] = f2bf(r1);
  op[lane + 32] = f2bf(r2);
  float ss = r1 * r1 + r2 * r2;
#pragma unroll
  for (int m = 16; m; m >>= 1) ss += __shfl_xor(ss, m, 32);
  if (lane == 0) sq[gw] = 0.5f * ss;
}

// ------------------------------------------------- per-row max over MF (query stab)
__global__ __launch_bounds__(256) void rowmax_kernel(const float* __restrict__ xp,
                                                     float* __restrict__ stab,
                                                     long long rows) {
  long long gw = ((long long)blockIdx.x * blockDim.x + threadIdx.x) >> 5;
  int lane = threadIdx.x & 31;
  if (gw >= rows) return;
  const f32x4* p4 = (const f32x4*)(xp + gw * MF);
  f32x4 v0 = p4[lane];
  f32x4 v1 = p4[lane + 32];
  float m = fmaxf(fmaxf(fmaxf(v0[0], v0[1]), fmaxf(v0[2], v0[3])),
                  fmaxf(fmaxf(v1[0], v1[1]), fmaxf(v1[2], v1[3])));
#pragma unroll
  for (int s = 16; s; s >>= 1) m = fmaxf(m, __shfl_xor(m, s, 32));
  if (lane == 0) stab[gw] = m;
}

// ------------------------------------------------- per-(b,kvh) max over (s,m)
__global__ __launch_bounds__(256) void headmax_kernel(const float* __restrict__ xpk,
                                                      float* __restrict__ stab) {
  int hb = blockIdx.x;  // b*8+kvh
  int b = hb >> 3, kvh = hb & 7;
  float m = -1e30f;
  for (long long idx = threadIdx.x; idx < (long long)SS * MF / 4; idx += blockDim.x) {
    long long s = idx >> 6;
    int m4 = (int)(idx & 63);
    f32x4 v = *(const f32x4*)(xpk + (((long long)b * SS + s) * NKVV + kvh) * MF + m4 * 4);
    m = fmaxf(m, fmaxf(fmaxf(v[0], v[1]), fmaxf(v[2], v[3])));
  }
  __shared__ float red[256];
  red[threadIdx.x] = m;
  __syncthreads();
  for (int st = 128; st; st >>= 1) {
    if ((int)threadIdx.x < st) red[threadIdx.x] = fmaxf(red[threadIdx.x], red[threadIdx.x + st]);
    __syncthreads();
  }
  if (threadIdx.x == 0) stab[hb] = red[0];
}

// ------------------------------------------------- phi = (exp(xp-sq-stab)+eps)/16
__global__ __launch_bounds__(256) void phi_q_kernel(const float* __restrict__ xp,
                                                    const float* __restrict__ sq,
                                                    const float* __restrict__ stab,
                                                    __bf16* __restrict__ phi) {
  long long n4 = (long long)QROWS * MF / 4;
  long long i4 = (long long)blockIdx.x * blockDim.x + threadIdx.x;
  long long stride = (long long)gridDim.x * blockDim.x;
  for (; i4 < n4; i4 += stride) {
    long long i = i4 * 4;
    long long r = i >> 8;
    float sub = sq[r] + stab[r];
    f32x4 x = *(const f32x4*)(xp + i);
    union { u16x4 v; __bf16 e[4]; } o;
#pragma unroll
    for (int j = 0; j < 4; ++j) o.e[j] = f2bf((expf(x[j] - sub) + EPSF) * 0.0625f);
    *(u16x4*)(phi + i) = o.v;
  }
}
__global__ __launch_bounds__(256) void phi_k_kernel(const float* __restrict__ xp,
                                                    const float* __restrict__ sq,
                                                    const float* __restrict__ stabh,
                                                    __bf16* __restrict__ phi) {
  long long n4 = (long long)KROWS * MF / 4;
  long long i4 = (long long)blockIdx.x * blockDim.x + threadIdx.x;
  long long stride = (long long)gridDim.x * blockDim.x;
  for (; i4 < n4; i4 += stride) {
    long long i = i4 * 4;
    long long r = i >> 8;              // (b*S+s)*8+kvh
    int kvh = (int)(r & 7);
    int b = (int)(r >> 14);            // / (S*8)
    float sub = sq[r] + stabh[b * NKVV + kvh];
    f32x4 x = *(const f32x4*)(xp + i);
    union { u16x4 v; __bf16 e[4]; } o;
#pragma unroll
    for (int j = 0; j < 4; ++j) o.e[j] = f2bf((expf(x[j] - sub) + EPSF) * 0.0625f);
    *(u16x4*)(phi + i) = o.v;
  }
}

// ------------------------------------------------- kz[b,kvh,c,m] = sum_t phi_k
__global__ __launch_bounds__(256) void kz_kernel(const __bf16* __restrict__ phik,
                                                 float* __restrict__ kz) {
  int idx = blockIdx.x * blockDim.x + threadIdx.x;
  if (idx >= NZK * MF) return;
  int m = idx & (MF - 1), c = (idx >> 8) & 15, kvh = (idx >> 12) & 7, b = idx >> 15;
  const __bf16* base = phik + (((long long)(b * SS + c * CH)) * NKVV + kvh) * MF + m;
  float s = 0.f;
  for (int t = 0; t < CH; ++t) s += bf2f(base[(long long)t * (NKVV * MF)]);
  kz[idx] = s;
}

// ------------------------------------------------- exclusive chunk scans
__global__ __launch_bounds__(256) void scan_kz(float* __restrict__ kz) {
  int idx = blockIdx.x * blockDim.x + threadIdx.x;  // (head, m): 16*256
  if (idx >= BB * NKVV * MF) return;
  int m = idx & (MF - 1), head = idx >> 8;
  float* base = kz + (long long)head * NC * MF + m;
  float acc = 0.f;
  for (int c = 0; c < NC; ++c) { float v = base[c * MF]; base[c * MF] = acc; acc += v; }
}
__global__ __launch_bounds__(256) void scan_kv(const float* __restrict__ kv,
                                               __bf16* __restrict__ kvpre) {
  long long idx4 = (long long)blockIdx.x * blockDim.x + threadIdx.x;  // (head, e/4)
  if (idx4 >= (long long)BB * NKVV * MF * DD / 4) return;
  long long e = (idx4 & (MF * DD / 4 - 1)) * 4;
  long long head = idx4 >> 12;
  const float* src = kv + head * NC * (MF * DD) + e;
  __bf16* dst = kvpre + head * NC * (MF * DD) + e;
  f32x4 acc = (f32x4)0.0f;
  for (int c = 0; c < NC; ++c) {
    f32x4 v = *(const f32x4*)(src + (long long)c * (MF * DD));
    union { u16x4 v; __bf16 e[4]; } o;
#pragma unroll
    for (int j = 0; j < 4; ++j) o.e[j] = f2bf(acc[j]);
    *(u16x4*)(dst + (long long)c * (MF * DD)) = o.v;
    acc += v;
  }
}

// ------------------------------------------------- causal mask + rowsum + bf16
__global__ __launch_bounds__(256) void mask_scores(const float* __restrict__ sc,
                                                   __bf16* __restrict__ scb,
                                                   float* __restrict__ intraz) {
  int idx = blockIdx.x * blockDim.x + threadIdx.x;  // (z, srow)
  if (idx >= NZ * CH) return;
  int srow = idx & (CH - 1);
  long long z = idx >> 7;
  const float* sp = sc + z * (CH * CH) + (long long)srow * CH;
  __bf16* op = scb + z * (CH * CH) + (long long)srow * CH;
  float sum = 0.f;
  for (int t = 0; t < CH; t += 4) {
    f32x4 v = *(const f32x4*)(sp + t);
    union { u16x4 v; __bf16 e[4]; } o;
#pragma unroll
    for (int j = 0; j < 4; ++j) {
      float x = (t + j <= srow) ? v[j] : 0.f;
      sum += x;
      o.e[j] = f2bf(x);
    }
    *(u16x4*)(op + t) = o.v;
  }
  intraz[idx] = sum;
}

// ------------------------------------------------- inter_z = qc . kz_pre
__global__ __launch_bounds__(256) void interz_kernel(const __bf16* __restrict__ phiq,
                                                     const float* __restrict__ kz,
                                                     float* __restrict__ interz) {
  int idx = blockIdx.x * blockDim.x + threadIdx.x;  // (z, srow)
  if (idx >= NZ * CH) return;
  int srow = idx & (CH - 1);
  int z = idx >> 7;
  int c = z & 15, h = (z >> 4) & 31, b = z >> 9;
  const __bf16* q = phiq + (((long long)(b * SS + c * CH + srow)) * NHH + h) * MF;
  const float* kzp = kz + (((long long)(b * NKVV + (h >> 2))) * NC + c) * MF;
  float s = 0.f;
  for (int m = 0; m < MF; m += 8) {
    union { v8bf v; __bf16 e[8]; } qv;
    qv.v = *(const v8bf*)(q + m);
    f32x4 z0 = *(const f32x4*)(kzp + m);
    f32x4 z1 = *(const f32x4*)(kzp + m + 4);
    s += bf2f(qv.e[0]) * z0[0] + bf2f(qv.e[1]) * z0[1] +
         bf2f(qv.e[2]) * z0[2] + bf2f(qv.e[3]) * z0[3] +
         bf2f(qv.e[4]) * z1[0] + bf2f(qv.e[5]) * z1[1] +
         bf2f(qv.e[6]) * z1[2] + bf2f(qv.e[7]) * z1[3];
  }
  interz[idx] = s;
}

// ------------------------------------------------- normalize + pack attn (bf16)
__global__ __launch_bounds__(256) void combine_kernel(const float* __restrict__ inter,
                                                      const float* __restrict__ intra,
                                                      const float* __restrict__ interz,
                                                      const float* __restrict__ intraz,
                                                      __bf16* __restrict__ attn) {
  long long n4 = (long long)NZ * CH * DD / 4;
  long long i4 = (long long)blockIdx.x * blockDim.x + threadIdx.x;
  long long stride = (long long)gridDim.x * blockDim.x;
  for (; i4 < n4; i4 += stride) {
    long long i = i4 * 4;
    int d = (int)(i & (DD - 1));        // multiple of 4, same (z,srow) for all 4
    long long t = i >> 6;               // (z, srow)
    int srow = (int)(t & (CH - 1));
    int z = (int)(t >> 7);
    int c = z & 15, h = (z >> 4) & 31, b = z >> 9;
    f32x4 a = *(const f32x4*)(inter + i);
    f32x4 q = *(const f32x4*)(intra + i);
    float den = 1.0f / (interz[t] + intraz[t] + EPSF);
    union { u16x4 v; __bf16 e[4]; } o;
#pragma unroll
    for (int j = 0; j < 4; ++j) o.e[j] = f2bf((a[j] + q[j]) * den);
    long long row = (long long)b * SS + c * CH + srow;
    *(u16x4*)(attn + row * (NHH * DD) + h * DD + d) = o.v;
  }
}

// =================================================================== launch
extern "C" void kernel_launch(void* const* d_in, const int* in_sizes, int n_in,
                              void* d_out, int out_size, void* d_ws, size_t ws_size,
                              hipStream_t stream) {
  const float* hs   = (const float*)d_in[0];
  const float* cosb = (const float*)d_in[1];
  const float* sinb = (const float*)d_in[2];
  const float* Wq   = (const float*)d_in[3];
  const float* Wk   = (const float*)d_in[4];
  const float* Wv   = (const float*)d_in[5];
  const float* Wo   = (const float*)d_in[6];
  const float* proj = (const float*)d_in[7];
  float* out = (float*)d_out;

  char* w = (char*)d_ws;
  auto alloc = [&](size_t bytes) -> char* {
    char* p = w;
    w += (bytes + 255) & ~(size_t)255;
    return p;
  };
  __bf16* hs_bf   = (__bf16*)alloc((size_t)ROWS * HIDD * 2);
  __bf16* Wq_bf   = (__bf16*)alloc((size_t)NHH * DD * HIDD * 2);
  __bf16* Wk_bf   = (__bf16*)alloc((size_t)NKVV * DD * HIDD * 2);
  __bf16* Wv_bf   = (__bf16*)alloc((size_t)NKVV * DD * HIDD * 2);
  __bf16* Wo_bf   = (__bf16*)alloc((size_t)HIDD * NHH * DD * 2);
  __bf16* proj_bf = (__bf16*)alloc((size_t)MF * DD * 2);
  float*  q_f     = (float*)alloc((size_t)ROWS * NHH * DD * 4);
  float*  k_f     = (float*)alloc((size_t)ROWS * NKVV * DD * 4);
  float*  v_f     = (float*)alloc((size_t)ROWS * NKVV * DD * 4);
  __bf16* v_bf    = (__bf16*)alloc((size_t)ROWS * NKVV * DD * 2);
  __bf16* qf_bf   = (__bf16*)alloc((size_t)ROWS * NHH * DD * 2);
  __bf16* kf_bf   = (__bf16*)alloc((size_t)ROWS * NKVV * DD * 2);
  float*  sq_q    = (float*)alloc((size_t)QROWS * 4);
  float*  sq_k    = (float*)alloc((size_t)KROWS * 4);
  float*  xp_q    = (float*)alloc((size_t)QROWS * MF * 4);
  float*  xp_k    = (float*)alloc((size_t)KROWS * MF * 4);
  float*  stab_q  = (float*)alloc((size_t)QROWS * 4);
  float*  stab_k  = (float*)alloc(256);
  __bf16* phi_q   = (__bf16*)alloc((size_t)QROWS * MF * 2);
  __bf16* phi_k   = (__bf16*)alloc((size_t)KROWS * MF * 2);
  float*  kv      = (float*)alloc((size_t)NZK * MF * DD * 4);
  __bf16* kv_pre  = (__bf16*)alloc((size_t)NZK * MF * DD * 2);
  float*  kz      = (float*)alloc((size_t)NZK * MF * 4);
  float*  inter   = (float*)alloc((size_t)NZ * CH * DD * 4);
  float*  scores  = (float*)alloc((size_t)NZ * CH * CH * 4);
  __bf16* scoresb = (__bf16*)alloc((size_t)NZ * CH * CH * 2);
  float*  intra   = (float*)alloc((size_t)NZ * CH * DD * 4);
  float*  interz  = (float*)alloc((size_t)NZ * CH * 4);
  float*  intraz  = (float*)alloc((size_t)NZ * CH * 4);
  __bf16* attn_bf = (__bf16*)alloc((size_t)ROWS * NHH * DD * 2);
  (void)alloc(1 << 20);  // tail pad so unconditional prefetch overruns stay mapped

  // 1) convert inputs to bf16 (vectorized x4)
  cvt_f32_bf16<<<2048, 256, 0, stream>>>(hs, hs_bf, (long long)ROWS * HIDD / 4);
  cvt_f32_bf16<<<2048, 256, 0, stream>>>(Wq, Wq_bf, (long long)NHH * DD * HIDD / 4);
  cvt_f32_bf16<<<1024, 256, 0, stream>>>(Wk, Wk_bf, (long long)NKVV * DD * HIDD / 4);
  cvt_f32_bf16<<<1024, 256, 0, stream>>>(Wv, Wv_bf, (long long)NKVV * DD * HIDD / 4);
  cvt_f32_bf16<<<2048, 256, 0, stream>>>(Wo, Wo_bf, (long long)HIDD * NHH * DD / 4);
  cvt_f32_bf16<<<16, 256, 0, stream>>>(proj, proj_bf, (long long)MF * DD / 4);

  // 2) QKV projections (A = hs [rows,K], B = W [N,K]); block 128x128
  gemm_bf16<false, false, 4><<<dim3(NHH * DD / 128, ROWS / 128, 1), 256, 0, stream>>>(
      hs_bf, Wq_bf, q_f, ROWS, NHH * DD, HIDD, HIDD, HIDD, NHH * DD, 0);
  gemm_bf16<false, false, 4><<<dim3(NKVV * DD / 128, ROWS / 128, 1), 256, 0, stream>>>(
      hs_bf, Wk_bf, k_f, ROWS, NKVV * DD, HIDD, HIDD, HIDD, NKVV * DD, 0);
  gemm_bf16<false, false, 4><<<dim3(NKVV * DD / 128, ROWS / 128, 1), 256, 0, stream>>>(
      hs_bf, Wv_bf, v_f, ROWS, NKVV * DD, HIDD, HIDD, HIDD, NKVV * DD, 0);
  cvt_f32_bf16<<<1024, 256, 0, stream>>>(v_f, v_bf, (long long)ROWS * NKVV * DD / 4);

  // 3) RoPE + d^-1/4 scale + 0.5||x||^2 (fused)
  rope_feat<<<(QROWS * 32) / 256, 256, 0, stream>>>(q_f, cosb, sinb, qf_bf, sq_q, ROWS, NHH);
  rope_feat<<<(KROWS * 32) / 256, 256, 0, stream>>>(k_f, cosb, sinb, kf_bf, sq_k, ROWS, NKVV);

  // 4) feature projections xp = xn @ proj^T
  gemm_bf16<false, false, 4><<<dim3(MF / 128, QROWS / 128, 1), 256, 0, stream>>>(
      qf_bf, proj_bf, xp_q, QROWS, MF, DD, DD, DD, MF, 0);
  gemm_bf16<false, false, 4><<<dim3(MF / 128, KROWS / 128, 1), 256, 0, stream>>>(
      kf_bf, proj_bf, xp_k, KROWS, MF, DD, DD, DD, MF, 0);

  // 5) stabilizers + phi
  rowmax_kernel<<<(QROWS * 32) / 256, 256, 0, stream>>>(xp_q, stab_q, (long long)QROWS);
  headmax_kernel<<<BB * NKVV, 256, 0, stream>>>(xp_k, stab_k);
  phi_q_kernel<<<8192, 256, 0, stream>>>(xp_q, sq_q, stab_q, phi_q);
  phi_k_kernel<<<4096, 256, 0, stream>>>(xp_k, sq_k, stab_k, phi_k);

  // 6) per-chunk KV state kv[m,d] = kc^T vc  (both operands K(t)-major)
  kz_kernel<<<(NZK * MF) / 256, 256, 0, stream>>>(phi_k, kz);
  gemm_bf16<true, true, 2><<<dim3(DD / 64, MF / 128, NZK), 256, 0, stream>>>(
      phi_k, v_bf, kv, MF, DD, CH, NKVV * MF, NKVV * DD, DD, 1);

  // 7) exclusive prefix over chunks
  scan_kz<<<(BB * NKVV * MF) / 256, 256, 0, stream>>>(kz);
  scan_kv<<<(BB * NKVV * MF * DD / 4) / 256, 256, 0, stream>>>(kv, kv_pre);

  // 8) inter = qc @ kv_pre ; scores = qc @ kc^T ; mask ; intra = scores @ vc
  gemm_bf16<false, true, 2><<<dim3(1, 1, NZ), 256, 0, stream>>>(
      phi_q, kv_pre, inter, CH, DD, MF, NHH * MF, DD, DD, 2);
  gemm_bf16<false, false, 4><<<dim3(CH / 128, 1, NZ), 256, 0, stream>>>(
      phi_q, phi_k, scores, CH, CH, MF, NHH * MF, NKVV * MF, CH, 3);
  mask_scores<<<(NZ * CH) / 256, 256, 0, stream>>>(scores, scoresb, intraz);
  gemm_bf16<false, true, 2><<<dim3(1, 1, NZ), 256, 0, stream>>>(
      scoresb, v_bf, intra, CH, DD, CH, CH, NKVV * DD, DD, 4);
  interz_kernel<<<(NZ * CH) / 256, 256, 0, stream>>>(phi_q, kz, interz);

  // 9) normalize, pack heads, output projection
  combine_kernel<<<8192, 256, 0, stream>>>(inter, intra, interz, intraz, attn_bf);
  gemm_bf16<false, false, 4><<<dim3(HIDD / 128, ROWS / 128, 1), 256, 0, stream>>>(
      attn_bf, Wo_bf, out, ROWS, HIDD, NHH * DD, NHH * DD, NHH * DD, HIDD, 0);

  (void)in_sizes; (void)n_in; (void)out_size; (void)ws_size;
}
